// ResnetBlock_21723944583655
// MI455X (gfx1250) — compile-verified, matrix operand order checked by reading, not run on hardware
//
#include <hip/hip_runtime.h>

// ---------------------------------------------------------------------------
// KPConv ResNet block for MI455X (gfx1250, wave32, WMMA f16->f32)
//
// Pipeline:
//   pack(W0), pack(W1), pack(W_short)        -> f16 WMMA-fragment layout
//   zero(stats)
//   kpconv<64,false>(x)        -> h1_raw [N,128] f32 + BN0 sums
//   bn_finalize(BN0)           -> scale0/shift0
//   kpconv<128,true>(h1_raw)   -> h2_raw (BN0+leaky applied on gather) + BN1 sums
//   bn_finalize(BN1)           -> scale1/shift1
//   final: out = leaky(h2_raw*scale1+shift1) + WMMA(x @ W_short)
// ---------------------------------------------------------------------------

typedef _Float16 h16 __attribute__((ext_vector_type(16)));
typedef _Float16 h8  __attribute__((ext_vector_type(8)));
typedef _Float16 h4  __attribute__((ext_vector_type(4)));
typedef float    f8  __attribute__((ext_vector_type(8)));
typedef float    f4  __attribute__((ext_vector_type(4)));

#define KP_EXTENT_INV (1.0f / 1.2f)
#define NEG_SLOPE 0.2f
#define BN_EPS 1e-5f

// 16-bit A/B fragment element mapping for V_WMMA_*_16X16X32 (ISA 7.12.2):
// lane sub-half sub = lane>>4; element i -> VGPR v=i>>1, half lo=i&1
// k = (v<4 ? 0 : 16) + sub*8 + (v%4)*2 + lo
__device__ __forceinline__ int kmap(int lane, int i) {
  int sub = lane >> 4;
  int v = i >> 1;
  return ((v & 4) << 2) + sub * 8 + ((v & 3) << 1) + (i & 1);
}

__device__ __forceinline__ h16 zero16() {
  h16 z;
#pragma unroll
  for (int q = 0; q < 16; ++q) z[q] = (_Float16)0.0f;
  return z;
}
__device__ __forceinline__ f8 zero8() {
  f8 z;
#pragma unroll
  for (int q = 0; q < 8; ++q) z[q] = 0.0f;
  return z;
}

// ---------------------------------------------------------------------------
// Pack a row-major f32 weight matrix W[rows=ksteps*32][128] into f16 WMMA
// B-fragments: Wp[((ks*8 + ntile)*32 + lane)*16 + i] = W[ks*32+kmap][ntile*16+col]
// ---------------------------------------------------------------------------
__global__ void pack_w_kernel(const float* __restrict__ W, _Float16* __restrict__ Wp,
                              int ksteps) {
  int tid = blockIdx.x * blockDim.x + threadIdx.x;
  int total = ksteps << 12; // ksteps * 8 * 32 * 16
  if (tid >= total) return;
  int i    = tid & 15;
  int lane = (tid >> 4) & 31;
  int t    = (tid >> 9) & 7;
  int ks   = tid >> 12;
  int row  = ks * 32 + kmap(lane, i);
  int col  = t * 16 + (lane & 15);
  Wp[tid] = (_Float16)W[row * 128 + col];
}

__global__ void zero_kernel(float* p, int n) {
  int i = blockIdx.x * blockDim.x + threadIdx.x;
  if (i < n) p[i] = 0.0f;
}

// ---------------------------------------------------------------------------
// KPConv: one 16-point tile per wave, 2 waves per block (64 threads).
// Output columns = 128 -> 8 WMMA accumulator tiles per wave.
// ---------------------------------------------------------------------------
template <int CIN, bool APPLY_BN>
__global__ __launch_bounds__(64) void kpconv_kernel(
    const float* __restrict__ pts, const int* __restrict__ neigh,
    const float* __restrict__ feats, const float* __restrict__ kp,
    const _Float16* __restrict__ Wp,
    const float* __restrict__ bn_scale, const float* __restrict__ bn_shift,
    float* __restrict__ out_raw, float* __restrict__ sums,
    float* __restrict__ sumsq, int N) {
  static_assert(CIN % 32 == 0, "CIN must be multiple of 32");
  extern __shared__ char smem[];
  const int wave = threadIdx.x >> 5;
  const int lane = threadIdx.x & 31;

  // per-wave LDS regions (2 waves)
  _Float16* lds_nf   = (_Float16*)smem + wave * (16 * 16 * 32);              // 16KB/wave
  _Float16* lds_infl = (_Float16*)(smem + 2 * 16 * 16 * 32 * 2) + wave * (16 * 16 * 16); // 8KB/wave
  int* lds_nbr = (int*)(smem + 2 * 16 * 16 * 32 * 2 + 2 * 16 * 16 * 16 * 2) + wave * 256;
  float* lds_bs = (float*)(smem + 2 * 16 * 16 * 32 * 2 + 2 * 16 * 16 * 16 * 2 + 2 * 256 * 4); // 256 f32 block stats

  for (int c = threadIdx.x; c < 256; c += 64) lds_bs[c] = 0.0f;

  const int tile = blockIdx.x * 2 + wave;
  const int m0 = tile * 16;

  // hoist kernel points into registers
  float kpx[15], kpy[15], kpz[15];
#pragma unroll
  for (int k = 0; k < 15; ++k) {
    kpx[k] = kp[k * 3 + 0];
    kpy[k] = kp[k * 3 + 1];
    kpz[k] = kp[k * 3 + 2];
  }

  // Phase A: neighbor indices + kernel-point influences -> LDS (f16)
#pragma unroll
  for (int j = 0; j < 8; ++j) {
    int idx = j * 32 + lane;
    int p = idx >> 4, e = idx & 15;
    int nb = neigh[(m0 + p) * 16 + e];
    lds_nbr[idx] = nb;
    float qx = pts[(m0 + p) * 3 + 0];
    float qy = pts[(m0 + p) * 3 + 1];
    float qz = pts[(m0 + p) * 3 + 2];
    float sx = pts[nb * 3 + 0] - qx;
    float sy = pts[nb * 3 + 1] - qy;
    float sz = pts[nb * 3 + 2] - qz;
#pragma unroll
    for (int k = 0; k < 15; ++k) {
      float dx = sx - kpx[k], dy = sy - kpy[k], dz = sz - kpz[k];
      float d = sqrtf(dx * dx + dy * dy + dz * dz);
      float infl = 1.0f - d * KP_EXTENT_INV;
      infl = infl > 0.0f ? infl : 0.0f;
      lds_infl[idx * 16 + k] = (_Float16)infl;
    }
  }
  __syncthreads();

  f8 acc[8];
#pragma unroll
  for (int t = 0; t < 8; ++t) acc[t] = zero8();

  const int NCH = CIN / 32;
  const int p = lane & 15;
  const int sub = lane >> 4;
  const int grow = lane >> 3;       // gather: row-in-group 0..3
  const int gcq  = (lane & 7) * 4;  // gather: 4-channel sub-block

  for (int ch = 0; ch < NCH; ++ch) {
    // Phase B: gather neighbor features for this 32-channel chunk.
    // 8 lanes cover one 128B feature row (float4/lane, b128 coalesced);
    // 4 rows per iteration. BN0 + leaky applied on the fly for conv2.
    f4 scv, shv;
    if (APPLY_BN) {
      scv = *(const f4*)(bn_scale + ch * 32 + gcq);
      shv = *(const f4*)(bn_shift + ch * 32 + gcq);
    }
    for (int it = 0; it < 64; ++it) {
      int pe = it * 4 + grow;
      int nb = lds_nbr[pe];
      f4 v = *(const f4*)(feats + (size_t)nb * CIN + ch * 32 + gcq);  // global_load_b128
      h4 o;
#pragma unroll
      for (int q = 0; q < 4; ++q) {
        float y = v[q];
        if (APPLY_BN) {
          y = y * scv[q] + shv[q];
          y = y > 0.0f ? y : NEG_SLOPE * y;
        }
        o[q] = (_Float16)y;
      }
      *(h4*)(lds_nf + pe * 32 + gcq) = o;                             // ds_store_b64
    }
    __syncthreads();

    // Phase C: build wf A-fragments (packed f16 FMAs, co-executes with WMMA)
    // and multiply-accumulate against packed W fragments.
    for (int kg = 0; kg < 5; ++kg) {   // 15 kernel points, groups of 3
      // warm L2->L0 path for the next k-group's B fragments
      if (kg < 4) {
        int ks_n = ((kg + 1) * 3) * NCH + ch;
        __builtin_prefetch(((const h16*)Wp) + (size_t)(ks_n * 8) * 32 + lane, 0, 0);
      }
      h16 wf[3];
#pragma unroll
      for (int kk = 0; kk < 3; ++kk) wf[kk] = zero16();

      for (int e = 0; e < 16; ++e) {
        const _Float16* base = lds_nf + (p * 16 + e) * 32;
        h8 lo = *(const h8*)(base + sub * 8);         // ds_load_b128
        h8 hi = *(const h8*)(base + 16 + sub * 8);    // ds_load_b128
        h16 nf = __builtin_shufflevector(lo, hi, 0, 1, 2, 3, 4, 5, 6, 7, 8, 9,
                                         10, 11, 12, 13, 14, 15);
#pragma unroll
        for (int kk = 0; kk < 3; ++kk) {
          _Float16 w = lds_infl[(p * 16 + e) * 16 + (kg * 3 + kk)];
          h16 ws = {w, w, w, w, w, w, w, w, w, w, w, w, w, w, w, w};
          wf[kk] = nf * ws + wf[kk];                  // v_pk_fma_f16 x8
        }
      }

#pragma unroll
      for (int kk = 0; kk < 3; ++kk) {
        int kstep = (kg * 3 + kk) * NCH + ch;         // kc = k*CIN + ch*32
        const h16* B = ((const h16*)Wp) + (size_t)(kstep * 8) * 32;
#pragma unroll
        for (int t = 0; t < 8; ++t) {
          h16 b = B[t * 32 + lane];                   // global_load_b128 x2
          acc[t] = __builtin_amdgcn_wmma_f32_16x16x32_f16(
              false, wf[kk], false, b, (short)0, acc[t], false, false);
        }
      }
    }
    __syncthreads();
  }

  // Phase D: store raw output (f32) and per-channel sum / sum-of-squares.
#pragma unroll
  for (int t = 0; t < 8; ++t) {
    int col = t * 16 + p;
    float s = 0.0f, q = 0.0f;
#pragma unroll
    for (int v = 0; v < 8; ++v) {
      int row = sub * 8 + v;                          // C-layout: M = v + 8*sub
      float val = acc[t][v];
      out_raw[(size_t)(m0 + row) * 128 + col] = val;
      s += val;
      q += val * val;
    }
    s += __shfl_xor(s, 16, 32);
    q += __shfl_xor(q, 16, 32);
    if (sub == 0) {
      atomicAdd(&lds_bs[col], s);                     // ds_add_f32
      atomicAdd(&lds_bs[128 + col], q);
    }
  }
  __syncthreads();
  for (int c = threadIdx.x; c < 128; c += 64) {
    atomicAdd(&sums[c], lds_bs[c]);                   // global_atomic_add_f32
    atomicAdd(&sumsq[c], lds_bs[128 + c]);
  }
}

// ---------------------------------------------------------------------------
// Fold BN statistics + gamma/beta into per-channel scale/shift.
// ---------------------------------------------------------------------------
__global__ void bn_finalize_kernel(const float* __restrict__ sums,
                                   const float* __restrict__ sumsq,
                                   const float* __restrict__ gamma,
                                   const float* __restrict__ beta,
                                   float* __restrict__ scale,
                                   float* __restrict__ shift, int N) {
  int c = threadIdx.x;
  float inv_n = 1.0f / (float)N;
  float mu = sums[c] * inv_n;
  float var = sumsq[c] * inv_n - mu * mu;
  float rs = rsqrtf(var + BN_EPS);
  float sc = gamma[c] * rs;
  scale[c] = sc;
  shift[c] = beta[c] - mu * sc;
}

// ---------------------------------------------------------------------------
// Final: out = leaky(h2_raw*scale1+shift1) + x @ W_short (shortcut via WMMA).
// 4 waves per block, one 16-point tile per wave.
// ---------------------------------------------------------------------------
__global__ __launch_bounds__(128) void final_kernel(
    const float* __restrict__ x, const _Float16* __restrict__ Wps,
    const float* __restrict__ h2, const float* __restrict__ scale,
    const float* __restrict__ shift, float* __restrict__ out, int N) {
  extern __shared__ char smem[];
  const int wave = threadIdx.x >> 5;
  const int lane = threadIdx.x & 31;
  _Float16* lx = (_Float16*)smem + wave * (16 * 64);

  int tile = blockIdx.x * 4 + wave;
  int m0 = tile * 16;

  // stage x tile [16 x 64] as f16 in LDS (float4 loads, b64 LDS stores)
#pragma unroll
  for (int j = 0; j < 8; ++j) {
    int idx = (j * 32 + lane) * 4;   // f32 index, 0..1023
    f4 v = *(const f4*)(x + (size_t)m0 * 64 + idx);
    h4 o;
#pragma unroll
    for (int q = 0; q < 4; ++q) o[q] = (_Float16)v[q];
    *(h4*)(lx + idx) = o;
  }
  __syncthreads();

  const int p = lane & 15;
  const int sub = lane >> 4;
  f8 acc[8];
#pragma unroll
  for (int t = 0; t < 8; ++t) acc[t] = zero8();

#pragma unroll
  for (int ch = 0; ch < 2; ++ch) {  // Cin = 64 -> 2 k-steps of 32
    const _Float16* base = lx + p * 64 + ch * 32;
    h8 lo = *(const h8*)(base + sub * 8);
    h8 hi = *(const h8*)(base + 16 + sub * 8);
    h16 a = __builtin_shufflevector(lo, hi, 0, 1, 2, 3, 4, 5, 6, 7, 8, 9, 10,
                                    11, 12, 13, 14, 15);
    const h16* B = ((const h16*)Wps) + (size_t)(ch * 8) * 32;
#pragma unroll
    for (int t = 0; t < 8; ++t) {
      h16 b = B[t * 32 + lane];
      acc[t] = __builtin_amdgcn_wmma_f32_16x16x32_f16(false, a, false, b,
                                                      (short)0, acc[t], false,
                                                      false);
    }
  }

#pragma unroll
  for (int t = 0; t < 8; ++t) {
    int col = t * 16 + p;
    float sc = scale[col];
    float sh = shift[col];
#pragma unroll
    for (int v = 0; v < 8; ++v) {
      int row = sub * 8 + v;
      float hv = h2[(size_t)(m0 + row) * 128 + col];
      hv = hv * sc + sh;
      hv = hv > 0.0f ? hv : NEG_SLOPE * hv;
      out[(size_t)(m0 + row) * 128 + col] = hv + acc[t][v];
    }
  }
}

// ---------------------------------------------------------------------------
// Host launcher
// ---------------------------------------------------------------------------
extern "C" void kernel_launch(void* const* d_in, const int* in_sizes, int n_in,
                              void* d_out, int out_size, void* d_ws,
                              size_t ws_size, hipStream_t stream) {
  const float* points = (const float*)d_in[0];   // [N,3]
  const float* x      = (const float*)d_in[1];   // [N,64]
  const int*   neigh  = (const int*)d_in[2];     // [N,16]
  const float* kp     = (const float*)d_in[3];   // [15,3]
  const float* W0     = (const float*)d_in[4];   // [15,64,128]
  const float* W1     = (const float*)d_in[5];   // [15,128,128]
  const float* Wshort = (const float*)d_in[6];   // [64,128]
  const float* gamma0 = (const float*)d_in[7];
  const float* beta0  = (const float*)d_in[8];
  const float* gamma1 = (const float*)d_in[9];
  const float* beta1  = (const float*)d_in[10];
  float* out = (float*)d_out;

  const int N = in_sizes[0] / 3;                 // 65536

  // --- workspace carve ---
  char* ws = (char*)d_ws;
  size_t off = 0;
  auto carve = [&](size_t bytes) {
    void* ptr = ws + off;
    off = (off + bytes + 255) & ~(size_t)255;
    return ptr;
  };
  float*    h1    = (float*)carve((size_t)N * 128 * 4);
  float*    h2    = (float*)carve((size_t)N * 128 * 4);
  _Float16* wp0   = (_Float16*)carve((size_t)30 * 4096 * 2);
  _Float16* wp1   = (_Float16*)carve((size_t)60 * 4096 * 2);
  _Float16* wps   = (_Float16*)carve((size_t)2 * 4096 * 2);
  float*    stats = (float*)carve(512 * 4);   // sums0,sq0,sums1,sq1
  float*    bn0   = (float*)carve(256 * 4);   // scale0,shift0
  float*    bn1   = (float*)carve(256 * 4);   // scale1,shift1
  (void)ws_size;

  // --- weight packing into WMMA fragment layout (f16) ---
  pack_w_kernel<<<(30 * 4096 + 255) / 256, 256, 0, stream>>>(W0, wp0, 30);
  pack_w_kernel<<<(60 * 4096 + 255) / 256, 256, 0, stream>>>(W1, wp1, 60);
  pack_w_kernel<<<(2 * 4096 + 255) / 256, 256, 0, stream>>>(Wshort, wps, 2);
  zero_kernel<<<2, 256, 0, stream>>>(stats, 512);

  const size_t CONV_LDS = 2 * 16 * 16 * 32 * 2   // nf chunks (f16)
                        + 2 * 16 * 16 * 16 * 2   // influences (f16)
                        + 2 * 256 * 4            // neighbor indices
                        + 256 * 4;               // block BN partials
  const int conv_blocks = N / 32;                // 2 tiles (waves) per block

  // conv1: x [N,64] -> h1_raw + BN0 stats (no input BN)
  kpconv_kernel<64, false><<<conv_blocks, 64, CONV_LDS, stream>>>(
      points, neigh, x, kp, wp0, nullptr, nullptr, h1, stats + 0, stats + 128, N);
  bn_finalize_kernel<<<1, 128, 0, stream>>>(stats + 0, stats + 128, gamma0,
                                            beta0, bn0, bn0 + 128, N);

  // conv2: BN0+leaky applied on gather of h1_raw -> h2_raw + BN1 stats
  kpconv_kernel<128, true><<<conv_blocks, 64, CONV_LDS, stream>>>(
      points, neigh, h1, kp, wp1, bn0, bn0 + 128, h2, stats + 256, stats + 384, N);
  bn_finalize_kernel<<<1, 128, 0, stream>>>(stats + 256, stats + 384, gamma1,
                                            beta1, bn1, bn1 + 128, N);

  // final: BN1+leaky on h2_raw + shortcut WMMA GEMM
  const size_t FIN_LDS = 4 * 16 * 64 * 2;
  final_kernel<<<N / 64, 128, FIN_LDS, stream>>>(x, wps, h2, bn1, bn1 + 128,
                                                 out, N);
}
